// Processor_35201551958352
// MI455X (gfx1250) — matrix operand order repaired and checked by reading, not compiled, not run on hardware
//
#include <hip/hip_runtime.h>
#include <hip/hip_bf16.h>
#include <math.h>

// ---------------------------------------------------------------------------
// Problem constants (match the reference)
// ---------------------------------------------------------------------------
#define NNODES   30000
#define NEDGES   240000     // E = 2*E2, multiple of 32 -> no guards in edge kernels
#define HDIM     128
#define NLAYERS  2

typedef __attribute__((ext_vector_type(8)))  float   v8f;
typedef __attribute__((ext_vector_type(8)))  __bf16  v8bf;
typedef __attribute__((ext_vector_type(16))) __bf16  v16bf;

__device__ __forceinline__ __bf16 f2bf(float f) {
  union { float f; unsigned u; } v; v.f = f;
  unsigned r = v.u + 0x7FFFu + ((v.u >> 16) & 1u);   // round-to-nearest-even
  union { unsigned short u; __bf16 b; } o; o.u = (unsigned short)(r >> 16);
  return o.b;
}

__device__ __forceinline__ float gelu_exact(float x) {
  return 0.5f * x * (1.0f + erff(x * 0.70710678118654752f));
}

// 16x16x32 bf16 A-fragment (contiguous K order, same as B) from a row-major LDS row:
// lane<16 holds K {kb..kb+15} of row lane&15; lane>=16 holds K {kb+16..kb+31}.
__device__ __forceinline__ v16bf afrag(const __bf16* __restrict__ arow, int kb, int hi) {
  const __bf16* ap = arow + kb + hi * 16;
  v8bf a0 = *(const v8bf*)(ap);
  v8bf a1 = *(const v8bf*)(ap + 8);
  return __builtin_shufflevector(a0, a1, 0,1,2,3,4,5,6,7,8,9,10,11,12,13,14,15);
}

// Async DMA of 16 bytes/lane from global into LDS (ASYNCcnt-tracked).
// LDS operand is the byte offset = low 32 bits of the generic LDS address
// (ISA 10.2: LDS aperture truncates to addr[31:0]).
__device__ __forceinline__ void async_ld_b128(const void* gptr, void* lptr) {
  unsigned lds = (unsigned)(unsigned long long)lptr;
  asm volatile("global_load_async_to_lds_b128 %0, %1, off"
               :: "v"(lds), "v"(gptr) : "memory");
}
__device__ __forceinline__ void wait_async0() {
  asm volatile("s_wait_asynccnt 0" ::: "memory");
}

// ---------------------------------------------------------------------------
// Weight convert + transpose: W[K][128] f32 -> Wt[128][K] bf16 (N-major, so a
// lane's B-fragment = 16 contiguous K values of one output column)
// ---------------------------------------------------------------------------
__global__ void k_transpose_bf16(const float* __restrict__ W, __bf16* __restrict__ Wt, int K) {
  int idx = blockIdx.x * blockDim.x + threadIdx.x;
  if (idx >= K * HDIM) return;
  int k = idx / HDIM, n = idx - k * HDIM;
  Wt[(long)n * K + k] = f2bf(W[idx]);
}

// ---------------------------------------------------------------------------
// Degree count: deg[row[e]] += 1
// ---------------------------------------------------------------------------
__global__ void k_deg(const int* __restrict__ row, float* __restrict__ deg) {
  int e = blockIdx.x * blockDim.x + threadIdx.x;
  if (e < NEDGES) atomicAdd(&deg[row[e]], 1.0f);
}

// ---------------------------------------------------------------------------
// Shared MLP epilogue helpers are inlined per kernel; each kernel is a single
// wave (32 threads) owning a 32-row tile = two 16x16 C tiles per WMMA column,
// so every B fragment load feeds TWO v_wmma issues.
// ---------------------------------------------------------------------------

// Edge message: e_out = LN(W2 * gelu(W1 * [x_i | x_j | ea] + b1) + b2)
__global__ void __launch_bounds__(32) k_edge_msg(
    const float* __restrict__ x, const float* __restrict__ ea,
    const int* __restrict__ row, const int* __restrict__ col,
    const __bf16* __restrict__ W1t, const float* __restrict__ b1,
    const __bf16* __restrict__ W2t, const float* __restrict__ b2,
    const float* __restrict__ g, const float* __restrict__ bln,
    float* __restrict__ e_out) {
  __shared__ __align__(32) __bf16 Ab[32 * 384];
  __shared__ __align__(32) __bf16 Hb[32 * 128];
  __shared__ __align__(32) float  Ft[32 * 128];
  __shared__ float smu[32], srs[32];

  const int  tid  = threadIdx.x;
  const long tile = (long)blockIdx.x * 32;

  __builtin_prefetch(&ea[(tile + 32) * 128], 0, 1);   // next tile's edge stream

  // Stage A = [x_i | x_j | ea] as bf16
  for (int idx = tid; idx < 32 * 384; idx += 32) {
    int r = idx / 384, c = idx - r * 384;
    long e = tile + r;
    float v;
    if (c < 128)      v = x[(long)row[e] * 128 + c];
    else if (c < 256) v = x[(long)col[e] * 128 + (c - 128)];
    else              v = ea[e * 128 + (c - 256)];
    Ab[r * 384 + c] = f2bf(v);
  }
  __syncthreads();

  const int hi = tid >> 4;
  const int lc = tid & 15;
  const __bf16* arow0 = &Ab[lc * 384];
  const __bf16* arow1 = &Ab[(16 + lc) * 384];

  // Linear1 (K=384) + GELU -> Hb (bf16)
  for (int nt = 0; nt < 8; ++nt) {
    v8f acc0 = {}, acc1 = {};
    const __bf16* bcol = &W1t[(nt * 16 + lc) * 384];
    #pragma unroll
    for (int ks = 0; ks < 12; ++ks) {
      const int kb = ks * 32;
      v16bf b = *(const v16bf*)(bcol + kb + hi * 16);
      acc0 = __builtin_amdgcn_wmma_f32_16x16x32_bf16(false, afrag(arow0, kb, hi), false, b, (short)0, acc0, false, false);
      acc1 = __builtin_amdgcn_wmma_f32_16x16x32_bf16(false, afrag(arow1, kb, hi), false, b, (short)0, acc1, false, false);
    }
    const int gcol = nt * 16 + lc;
    const float bia = b1[gcol];
    #pragma unroll
    for (int i = 0; i < 8; ++i) {
      Hb[(hi * 8 + i) * 128 + gcol]        = f2bf(gelu_exact(acc0[i] + bia));
      Hb[(16 + hi * 8 + i) * 128 + gcol]   = f2bf(gelu_exact(acc1[i] + bia));
    }
  }
  __syncthreads();

  // Linear2 (K=128) -> Ft (f32)
  const __bf16* hrow0 = &Hb[lc * 128];
  const __bf16* hrow1 = &Hb[(16 + lc) * 128];
  for (int nt = 0; nt < 8; ++nt) {
    v8f acc0 = {}, acc1 = {};
    const __bf16* bcol = &W2t[(nt * 16 + lc) * 128];
    #pragma unroll
    for (int ks = 0; ks < 4; ++ks) {
      const int kb = ks * 32;
      v16bf b = *(const v16bf*)(bcol + kb + hi * 16);
      acc0 = __builtin_amdgcn_wmma_f32_16x16x32_bf16(false, afrag(hrow0, kb, hi), false, b, (short)0, acc0, false, false);
      acc1 = __builtin_amdgcn_wmma_f32_16x16x32_bf16(false, afrag(hrow1, kb, hi), false, b, (short)0, acc1, false, false);
    }
    const int gcol = nt * 16 + lc;
    const float bia = b2[gcol];
    #pragma unroll
    for (int i = 0; i < 8; ++i) {
      Ft[(hi * 8 + i) * 128 + gcol]      = acc0[i] + bia;
      Ft[(16 + hi * 8 + i) * 128 + gcol] = acc1[i] + bia;
    }
  }
  __syncthreads();

  // LayerNorm stats: one row per lane
  {
    float mu = 0.f;
    for (int c = 0; c < 128; ++c) mu += Ft[tid * 128 + c];
    mu *= (1.0f / 128.0f);
    float var = 0.f;
    for (int c = 0; c < 128; ++c) { float d = Ft[tid * 128 + c] - mu; var += d * d; }
    smu[tid] = mu;
    srs[tid] = rsqrtf(var * (1.0f / 128.0f) + 1e-5f);
  }
  __syncthreads();

  for (int idx = tid; idx < 32 * 128; idx += 32) {
    int r = idx >> 7, c = idx & 127;
    e_out[(tile + r) * 128 + c] = (Ft[idx] - smu[r]) * srs[r] * g[c] + bln[c];
  }
}

// ---------------------------------------------------------------------------
// Flux: flux = (e - e_rev) + LN(MLP(e + e_rev)); ea += flux; agg[row] += flux
// LDS arena (56KB), phase-overlaid:
//   phase1: Eb f32 [0,16K)   Rb f32 [16K,32K)       (async DMA staged)
//   phase2: Sb bf16 [32K,40K) Df f32 [40K,56K)
//   phase3: Hb bf16 [0,8K)    Ft f32 [16K,32K)      (Eb/Rb dead)
// ---------------------------------------------------------------------------
__global__ void __launch_bounds__(32) k_flux(
    const float* __restrict__ e_in, const int* __restrict__ rev,
    const int* __restrict__ row,
    const __bf16* __restrict__ W1t, const float* __restrict__ b1,
    const __bf16* __restrict__ W2t, const float* __restrict__ b2,
    const float* __restrict__ g, const float* __restrict__ bln,
    float* __restrict__ ea, float* __restrict__ agg) {
  __shared__ __align__(64) unsigned char Lds[57344];
  __shared__ float smu[32], srs[32];
  float*  Eb = (float*)(Lds);
  float*  Rb = (float*)(Lds + 16384);
  __bf16* Sb = (__bf16*)(Lds + 32768);
  float*  Df = (float*)(Lds + 40960);
  __bf16* Hb = (__bf16*)(Lds);
  float*  Ft = (float*)(Lds + 16384);

  const int  tid  = threadIdx.x;
  const long tile = (long)blockIdx.x * 32;

  __builtin_prefetch(&e_in[(tile + 32) * 128], 0, 1);

  // Phase 1: async DMA one 512B row per instruction (16B per lane)
  #pragma unroll 4
  for (int r = 0; r < 32; ++r) {
    long e = tile + r;
    async_ld_b128(&e_in[e * 128 + tid * 4],                Eb + r * 128 + tid * 4);
    async_ld_b128(&e_in[(long)rev[e] * 128 + tid * 4],     Rb + r * 128 + tid * 4);
  }
  wait_async0();
  __syncthreads();

  // Phase 2: sum (bf16, WMMA input) and difference (f32, antisymmetric part)
  for (int idx = tid; idx < 32 * 128; idx += 32) {
    float ev = Eb[idx], rv = Rb[idx];
    Sb[idx] = f2bf(ev + rv);
    Df[idx] = ev - rv;
  }
  __syncthreads();

  const int hi = tid >> 4;
  const int lc = tid & 15;

  const __bf16* arow0 = &Sb[lc * 128];
  const __bf16* arow1 = &Sb[(16 + lc) * 128];
  for (int nt = 0; nt < 8; ++nt) {
    v8f acc0 = {}, acc1 = {};
    const __bf16* bcol = &W1t[(nt * 16 + lc) * 128];
    #pragma unroll
    for (int ks = 0; ks < 4; ++ks) {
      const int kb = ks * 32;
      v16bf b = *(const v16bf*)(bcol + kb + hi * 16);
      acc0 = __builtin_amdgcn_wmma_f32_16x16x32_bf16(false, afrag(arow0, kb, hi), false, b, (short)0, acc0, false, false);
      acc1 = __builtin_amdgcn_wmma_f32_16x16x32_bf16(false, afrag(arow1, kb, hi), false, b, (short)0, acc1, false, false);
    }
    const int gcol = nt * 16 + lc;
    const float bia = b1[gcol];
    #pragma unroll
    for (int i = 0; i < 8; ++i) {
      Hb[(hi * 8 + i) * 128 + gcol]      = f2bf(gelu_exact(acc0[i] + bia));
      Hb[(16 + hi * 8 + i) * 128 + gcol] = f2bf(gelu_exact(acc1[i] + bia));
    }
  }
  __syncthreads();

  const __bf16* hrow0 = &Hb[lc * 128];
  const __bf16* hrow1 = &Hb[(16 + lc) * 128];
  for (int nt = 0; nt < 8; ++nt) {
    v8f acc0 = {}, acc1 = {};
    const __bf16* bcol = &W2t[(nt * 16 + lc) * 128];
    #pragma unroll
    for (int ks = 0; ks < 4; ++ks) {
      const int kb = ks * 32;
      v16bf b = *(const v16bf*)(bcol + kb + hi * 16);
      acc0 = __builtin_amdgcn_wmma_f32_16x16x32_bf16(false, afrag(hrow0, kb, hi), false, b, (short)0, acc0, false, false);
      acc1 = __builtin_amdgcn_wmma_f32_16x16x32_bf16(false, afrag(hrow1, kb, hi), false, b, (short)0, acc1, false, false);
    }
    const int gcol = nt * 16 + lc;
    const float bia = b2[gcol];
    #pragma unroll
    for (int i = 0; i < 8; ++i) {
      Ft[(hi * 8 + i) * 128 + gcol]      = acc0[i] + bia;
      Ft[(16 + hi * 8 + i) * 128 + gcol] = acc1[i] + bia;
    }
  }
  __syncthreads();

  {
    float mu = 0.f;
    for (int c = 0; c < 128; ++c) mu += Ft[tid * 128 + c];
    mu *= (1.0f / 128.0f);
    float var = 0.f;
    for (int c = 0; c < 128; ++c) { float d = Ft[tid * 128 + c] - mu; var += d * d; }
    smu[tid] = mu;
    srs[tid] = rsqrtf(var * (1.0f / 128.0f) + 1e-5f);
  }
  __syncthreads();

  for (int idx = tid; idx < 32 * 128; idx += 32) {
    int r = idx >> 7, c = idx & 127;
    long e = tile + r;
    float f = (Ft[idx] - smu[r]) * srs[r] * g[c] + bln[c] + Df[idx];
    ea[e * 128 + c] += f;                                  // edge residual
    atomicAdd(&agg[(long)row[e] * 128 + c], f);            // mean-agg numerator
  }
}

// ---------------------------------------------------------------------------
// Node update: x_out = x + LN(MLP([x | agg/deg]))    (32 nodes per block)
// ---------------------------------------------------------------------------
__global__ void __launch_bounds__(32) k_update(
    const float* __restrict__ xin, const float* __restrict__ agg,
    const float* __restrict__ deg,
    const __bf16* __restrict__ W1t, const float* __restrict__ b1,
    const __bf16* __restrict__ W2t, const float* __restrict__ b2,
    const float* __restrict__ g, const float* __restrict__ bln,
    float* __restrict__ xout) {
  __shared__ __align__(32) __bf16 Ab[32 * 256];
  __shared__ __align__(32) __bf16 Hb[32 * 128];
  __shared__ __align__(32) float  Ft[32 * 128];
  __shared__ float smu[32], srs[32];

  const int  tid  = threadIdx.x;
  const long tile = (long)blockIdx.x * 32;

  for (int idx = tid; idx < 32 * 256; idx += 32) {
    int r = idx >> 8, c = idx & 255;
    long n = tile + r; if (n >= NNODES) n = NNODES - 1;   // clamp (keeps EXEC full)
    float v;
    if (c < 128) v = xin[n * 128 + c];
    else         v = agg[n * 128 + (c - 128)] / fmaxf(deg[n], 1.0f);
    Ab[idx] = f2bf(v);
  }
  __syncthreads();

  const int hi = tid >> 4;
  const int lc = tid & 15;

  const __bf16* arow0 = &Ab[lc * 256];
  const __bf16* arow1 = &Ab[(16 + lc) * 256];
  for (int nt = 0; nt < 8; ++nt) {
    v8f acc0 = {}, acc1 = {};
    const __bf16* bcol = &W1t[(nt * 16 + lc) * 256];
    #pragma unroll
    for (int ks = 0; ks < 8; ++ks) {
      const int kb = ks * 32;
      v16bf b = *(const v16bf*)(bcol + kb + hi * 16);
      acc0 = __builtin_amdgcn_wmma_f32_16x16x32_bf16(false, afrag(arow0, kb, hi), false, b, (short)0, acc0, false, false);
      acc1 = __builtin_amdgcn_wmma_f32_16x16x32_bf16(false, afrag(arow1, kb, hi), false, b, (short)0, acc1, false, false);
    }
    const int gcol = nt * 16 + lc;
    const float bia = b1[gcol];
    #pragma unroll
    for (int i = 0; i < 8; ++i) {
      Hb[(hi * 8 + i) * 128 + gcol]      = f2bf(gelu_exact(acc0[i] + bia));
      Hb[(16 + hi * 8 + i) * 128 + gcol] = f2bf(gelu_exact(acc1[i] + bia));
    }
  }
  __syncthreads();

  const __bf16* hrow0 = &Hb[lc * 128];
  const __bf16* hrow1 = &Hb[(16 + lc) * 128];
  for (int nt = 0; nt < 8; ++nt) {
    v8f acc0 = {}, acc1 = {};
    const __bf16* bcol = &W2t[(nt * 16 + lc) * 128];
    #pragma unroll
    for (int ks = 0; ks < 4; ++ks) {
      const int kb = ks * 32;
      v16bf b = *(const v16bf*)(bcol + kb + hi * 16);
      acc0 = __builtin_amdgcn_wmma_f32_16x16x32_bf16(false, afrag(hrow0, kb, hi), false, b, (short)0, acc0, false, false);
      acc1 = __builtin_amdgcn_wmma_f32_16x16x32_bf16(false, afrag(hrow1, kb, hi), false, b, (short)0, acc1, false, false);
    }
    const int gcol = nt * 16 + lc;
    const float bia = b2[gcol];
    #pragma unroll
    for (int i = 0; i < 8; ++i) {
      Ft[(hi * 8 + i) * 128 + gcol]      = acc0[i] + bia;
      Ft[(16 + hi * 8 + i) * 128 + gcol] = acc1[i] + bia;
    }
  }
  __syncthreads();

  {
    float mu = 0.f;
    for (int c = 0; c < 128; ++c) mu += Ft[tid * 128 + c];
    mu *= (1.0f / 128.0f);
    float var = 0.f;
    for (int c = 0; c < 128; ++c) { float d = Ft[tid * 128 + c] - mu; var += d * d; }
    smu[tid] = mu;
    srs[tid] = rsqrtf(var * (1.0f / 128.0f) + 1e-5f);
  }
  __syncthreads();

  for (int idx = tid; idx < 32 * 128; idx += 32) {
    int r = idx >> 7, c = idx & 127;
    long n = tile + r;
    if (n < NNODES)
      xout[n * 128 + c] = xin[n * 128 + c]
                        + (Ft[idx] - smu[r]) * srs[r] * g[c] + bln[c];
  }
}

// ---------------------------------------------------------------------------
// Output CNN: pointwise over 3 channels -> 4 -> 1
// ---------------------------------------------------------------------------
__global__ void k_cnn(const float* __restrict__ x0, const float* __restrict__ x1,
                      const float* __restrict__ x2,
                      const float* __restrict__ w1, const float* __restrict__ b1,
                      const float* __restrict__ w2, const float* __restrict__ b2,
                      float* __restrict__ out) {
  int idx = blockIdx.x * blockDim.x + threadIdx.x;
  if (idx >= NNODES * HDIM) return;
  float v0 = x0[idx], v1 = x1[idx], v2 = x2[idx];
  float o = b2[0];
  #pragma unroll
  for (int c = 0; c < 4; ++c) {
    float t = gelu_exact(b1[c] + w1[c * 3 + 0] * v0 + w1[c * 3 + 1] * v1 + w1[c * 3 + 2] * v2);
    o += w2[c] * t;
  }
  out[idx] = o;
}

// ---------------------------------------------------------------------------
// Host
// ---------------------------------------------------------------------------
extern "C" void kernel_launch(void* const* d_in, const int* in_sizes, int n_in,
                              void* d_out, int out_size, void* d_ws, size_t ws_size,
                              hipStream_t stream) {
  (void)in_sizes; (void)n_in; (void)out_size; (void)ws_size;

  const float* x         = (const float*)d_in[0];
  const float* edge_attr = (const float*)d_in[1];
  const float* msg_w1    = (const float*)d_in[2];   // [L][384][128]
  const float* msg_b1    = (const float*)d_in[3];
  const float* msg_w2    = (const float*)d_in[4];   // [L][128][128]
  const float* msg_b2    = (const float*)d_in[5];
  const float* flux_w1   = (const float*)d_in[6];
  const float* flux_b1   = (const float*)d_in[7];
  const float* flux_w2   = (const float*)d_in[8];
  const float* flux_b2   = (const float*)d_in[9];
  const float* upd_w1    = (const float*)d_in[10];  // [L][256][128]
  const float* upd_b1    = (const float*)d_in[11];
  const float* upd_w2    = (const float*)d_in[12];
  const float* upd_b2    = (const float*)d_in[13];
  const float* ln_g      = (const float*)d_in[14];
  const float* ln_b      = (const float*)d_in[15];
  const float* cnn_w1    = (const float*)d_in[16];  // [4][3]
  const float* cnn_b1    = (const float*)d_in[17];
  const float* cnn_w2    = (const float*)d_in[18];  // [1][4]
  const float* cnn_b2    = (const float*)d_in[19];
  const int*   edge_idx  = (const int*)d_in[20];    // [2][E]
  const int*   rev_perm  = (const int*)d_in[21];    // [E]

  const int* row = edge_idx;
  const int* col = edge_idx + NEDGES;

  const size_t NH = (size_t)NNODES * HDIM;
  const size_t EH = (size_t)NEDGES * HDIM;

  // Workspace carve-up (256-byte aligned slices)
  unsigned char* p = (unsigned char*)d_ws;
  auto carve = [&](size_t bytes) -> void* {
    void* r = (void*)p;
    p += (bytes + 255) & ~(size_t)255;
    return r;
  };
  float*  xs    = (float*)carve(3 * NH * 4);       // x0, x1, x2
  float*  ea    = (float*)carve(EH * 4);
  float*  ebuf  = (float*)carve(EH * 4);
  float*  agg   = (float*)carve(NH * 4);
  float*  deg   = (float*)carve((size_t)NNODES * 4);
  __bf16* mW1t  = (__bf16*)carve(2 * 384 * 128 * 2);
  __bf16* mW2t  = (__bf16*)carve(2 * 128 * 128 * 2);
  __bf16* fW1t  = (__bf16*)carve(2 * 128 * 128 * 2);
  __bf16* fW2t  = (__bf16*)carve(2 * 128 * 128 * 2);
  __bf16* uW1t  = (__bf16*)carve(2 * 256 * 128 * 2);
  __bf16* uW2t  = (__bf16*)carve(2 * 128 * 128 * 2);

  // Init: copy residual streams, zero degree
  hipMemcpyAsync(xs, x, NH * 4, hipMemcpyDeviceToDevice, stream);
  hipMemcpyAsync(ea, edge_attr, EH * 4, hipMemcpyDeviceToDevice, stream);
  hipMemsetAsync(deg, 0, (size_t)NNODES * 4, stream);

  // Weight convert/transpose
  auto T = [&](const float* W, __bf16* Wt, int K) {
    int total = K * HDIM;
    k_transpose_bf16<<<(total + 255) / 256, 256, 0, stream>>>(W, Wt, K);
  };
  for (int l = 0; l < NLAYERS; ++l) {
    T(msg_w1  + (size_t)l * 384 * 128, mW1t + (size_t)l * 384 * 128, 384);
    T(msg_w2  + (size_t)l * 128 * 128, mW2t + (size_t)l * 128 * 128, 128);
    T(flux_w1 + (size_t)l * 128 * 128, fW1t + (size_t)l * 128 * 128, 128);
    T(flux_w2 + (size_t)l * 128 * 128, fW2t + (size_t)l * 128 * 128, 128);
    T(upd_w1  + (size_t)l * 256 * 128, uW1t + (size_t)l * 256 * 128, 256);
    T(upd_w2  + (size_t)l * 128 * 128, uW2t + (size_t)l * 128 * 128, 128);
  }

  k_deg<<<(NEDGES + 255) / 256, 256, 0, stream>>>(row, deg);

  const int edgeBlocks = NEDGES / 32;                 // exact: 7500
  const int nodeBlocks = (NNODES + 31) / 32;

  for (int l = 0; l < NLAYERS; ++l) {
    const float* xcur = xs + (size_t)l * NH;
    float*       xnxt = xs + (size_t)(l + 1) * NH;
    const size_t bo = (size_t)l * 128;

    hipMemsetAsync(agg, 0, NH * 4, stream);

    k_edge_msg<<<edgeBlocks, 32, 0, stream>>>(
        xcur, ea, row, col,
        mW1t + (size_t)l * 384 * 128, msg_b1 + bo,
        mW2t + (size_t)l * 128 * 128, msg_b2 + bo,
        ln_g + bo, ln_b + bo, ebuf);

    k_flux<<<edgeBlocks, 32, 0, stream>>>(
        ebuf, rev_perm, row,
        fW1t + (size_t)l * 128 * 128, flux_b1 + bo,
        fW2t + (size_t)l * 128 * 128, flux_b2 + bo,
        ln_g + bo, ln_b + bo, ea, agg);

    k_update<<<nodeBlocks, 32, 0, stream>>>(
        xcur, agg, deg,
        uW1t + (size_t)l * 256 * 128, upd_b1 + bo,
        uW2t + (size_t)l * 128 * 128, upd_b2 + bo,
        ln_g + bo, ln_b + bo, xnxt);
  }

  k_cnn<<<(int)((NH + 255) / 256), 256, 0, stream>>>(
      xs, xs + NH, xs + 2 * NH, cnn_w1, cnn_b1, cnn_w2, cnn_b2, (float*)d_out);
}
